// LongformerSelfAttentionForBart_76914274337234
// MI455X (gfx1250) — matrix operand order repaired and checked by reading, not compiled, not run on hardware
//
#include <hip/hip_runtime.h>

// ---------------------------------------------------------------------------
// Longformer self-attention (BART) for MI455X / gfx1250.
// wave32 + v_wmma_f32_16x16x32_bf16 + Tensor Data Mover double-buffering.
// B=2, S=4096, D=768, H=12, HD=64, W1=256, NC=16, G=8.
// ---------------------------------------------------------------------------

typedef __attribute__((ext_vector_type(16))) __bf16       v16bf;
typedef __attribute__((ext_vector_type(8)))  float        v8f;
typedef __attribute__((ext_vector_type(4)))  unsigned int u32x4;
typedef __attribute__((ext_vector_type(8)))  int          i32x8;
typedef __attribute__((ext_vector_type(4)))  int          i32x4;

#define DEV __device__ __forceinline__

#if defined(__AMDGCN__) && __has_builtin(__builtin_amdgcn_tensor_load_to_lds)
#define USE_TDM 1
#else
#define USE_TDM 0
#endif

constexpr int    Bc   = 2;
constexpr int    Sc   = 4096;
constexpr int    Dc   = 768;
constexpr int    Hc   = 12;
constexpr int    HDc  = 64;
constexpr int    W1c  = 256;
constexpr int    NCc  = 16;
constexpr int    Gc   = 8;
constexpr float  NEGF = -1.0e9f;
constexpr size_t BSDc = (size_t)Bc * Sc * Dc;   // 6291456 elements
constexpr size_t WNc  = (size_t)Dc * Dc;        // 589824 elements

// ---------------------------------------------------------------- WMMA helper
DEV v8f wmma_bf16(v16bf a, v16bf bm, v8f c) {
  return __builtin_amdgcn_wmma_f32_16x16x32_bf16(false, a, false, bm,
                                                 (short)0, c, false, false);
}

// A-fragment (16x32 bf16): lane m=lane&15, sel=lane>>4.
// elements 0..7 -> K = 8*sel+e ; elements 8..15 -> K = 16+8*sel+(e-8).
DEV v16bf load_a_frag(const __bf16* p, int ld) {
  int l = threadIdx.x & 31;
  int m = l & 15, sel = l >> 4;
  const __bf16* r = p + (size_t)m * ld + 8 * sel;
  v16bf a;
#pragma unroll
  for (int e = 0; e < 8; ++e) a[e] = r[e];
#pragma unroll
  for (int e = 0; e < 8; ++e) a[8 + e] = r[16 + e];
  return a;
}

// B-fragment (32x16, B[k][n]) from an [n][k] buffer: lane n=lane&15,
// sel = lane>>4, elements 0..7 -> K = 8*sel+e ; elements 8..15 -> K = 16+8*sel+(e-8).
DEV v16bf load_b_frag_nk(const __bf16* p, int ldk) {
  int l = threadIdx.x & 31;
  int n = l & 15, kh = l >> 4;
  const __bf16* r = p + (size_t)n * ldk + kh * 8;
  v16bf bm;
#pragma unroll
  for (int e = 0; e < 8; ++e) bm[e] = r[e];
#pragma unroll
  for (int e = 0; e < 8; ++e) bm[8 + e] = r[16 + e];
  return bm;
}

DEV float rowmax16(float x) {
  x = fmaxf(x, __shfl_xor(x, 1, 32));
  x = fmaxf(x, __shfl_xor(x, 2, 32));
  x = fmaxf(x, __shfl_xor(x, 4, 32));
  x = fmaxf(x, __shfl_xor(x, 8, 32));
  return x;
}
DEV float rowsum16(float x) {
  x += __shfl_xor(x, 1, 32);
  x += __shfl_xor(x, 2, 32);
  x += __shfl_xor(x, 4, 32);
  x += __shfl_xor(x, 8, 32);
  return x;
}

// Online-softmax update of two 16x16 score tiles; P written (bf16) to the
// wave's LDS scratch for re-load in A-fragment layout.
DEV void flash_update(v8f s0, v8f s1, float* m8, float* l8, v8f (&O)[4],
                      __bf16* ldsPw) {
  int l = threadIdx.x & 31;
  int m16 = l & 15, sel = l >> 4;
#pragma unroll
  for (int v = 0; v < 8; ++v) {
    float rm = rowmax16(fmaxf(s0[v], s1[v]));
    float mv = fmaxf(m8[v], rm);
    float fs = __expf(m8[v] - mv);
    float p0 = __expf(s0[v] - mv);
    float p1 = __expf(s1[v] - mv);
    float rs = rowsum16(p0 + p1);
    l8[v] = l8[v] * fs + rs;
    m8[v] = mv;
#pragma unroll
    for (int t = 0; t < 4; ++t) O[t][v] *= fs;
    int r = v + 8 * sel;
    ldsPw[r * 32 + m16]      = (__bf16)p0;
    ldsPw[r * 32 + 16 + m16] = (__bf16)p1;
  }
}

#if USE_TDM
// ------------------------------------------------- Tensor Data Mover (D#)
// 2-D tile load: tile_x elements per row (2B each), tile_y rows, rows
// separated by stride_x elements in global memory; packed contiguously in LDS.
// Descriptor bitfields per CDNA5 ISA.
// clang-23 toolchain builtin: 6 args (g0, g1, g2, g3, g-extra, cpol).
DEV void tdm_load_2d(const void* gaddr, unsigned lds_addr, unsigned tile_x,
                     unsigned tile_y, unsigned stride_x) {
  unsigned long long ga = (unsigned long long)gaddr;
  u32x4 g0;
  g0[0] = 1u;                                   // count=1, user mode
  g0[1] = lds_addr;                             // LDS byte address
  g0[2] = (unsigned)ga;                         // global_addr[31:0]
  g0[3] = (unsigned)((ga >> 32) & 0x01FFFFFFu)  // global_addr[56:32]
          | 0x80000000u;                        // type=2 ("image")
  i32x8 g1;
  g1[0] = (int)(1u << 16);                      // data_size=1 (2-byte elems)
  g1[1] = (int)((tile_x & 0xFFFFu) << 16);      // tensor_dim0[15:0] = tile_x
  g1[2] = (int)(((tile_x >> 16) & 0xFFFFu) |
                ((tile_y & 0xFFFFu) << 16));    // dim0[31:16] | dim1[15:0]
  g1[3] = (int)(((tile_y >> 16) & 0xFFFFu) |
                ((tile_x & 0xFFFFu) << 16));    // dim1[31:16] | tile_dim0
  g1[4] = (int)(tile_y & 0xFFFFu);              // tile_dim1 (tile_dim2=0)
  g1[5] = (int)stride_x;                        // tensor_dim0_stride[31:0]
  g1[6] = 0;
  g1[7] = 0;
  i32x4 z4 = {0, 0, 0, 0};
  i32x8 z8 = {0, 0, 0, 0, 0, 0, 0, 0};
  __builtin_amdgcn_tensor_load_to_lds(g0, g1, z4, z4, z8, 0);
}
DEV unsigned lds_off(const void* p) {           // generic->LDS byte offset
  return (unsigned)(unsigned long long)p;
}
#endif

// ------------------------------------------------------------ convert kernels
__global__ void cvt_bf16_kernel(const float* __restrict__ src,
                                __bf16* __restrict__ dst, int n) {
  int i = blockIdx.x * 256 + threadIdx.x;
  if (i < n) dst[i] = (__bf16)src[i];
}

// W [K][N] fp32 -> Wt [N][K] bf16
__global__ void transpose_bf16_kernel(const float* __restrict__ src,
                                      __bf16* __restrict__ dst) {
  int i = blockIdx.x * 256 + threadIdx.x;
  if (i < Dc * Dc) {
    int n = i / Dc, k = i % Dc;
    dst[i] = (__bf16)src[(size_t)k * Dc + n];
  }
}

// ------------------------------------------------------------------ GEMM
// C[8192][768] = A(bf16) * Wt^T + bias. Tile 128x64, 8 waves, K-step 64,
// TDM double-buffered LDS staging.
// mode 0: bf16 head layout [B,H,S,HD]; mode 1: bf16 [B,H,HD,S]; mode 2: f32.
__global__ __launch_bounds__(256) void gemm_bf16_kernel(
    const __bf16* __restrict__ A, const __bf16* __restrict__ Wt,
    const float* __restrict__ bias, void* __restrict__ outp, float scale,
    int mode) {
  __shared__ __bf16 ldsA[2][128 * 64];
  __shared__ __bf16 ldsB[2][64 * 64];

  int row0 = blockIdx.x * 128;
  int n0   = blockIdx.y * 64;
  int tid  = threadIdx.x;
  int wid  = tid >> 5, lane = tid & 31;
  int m16  = lane & 15, sel = lane >> 4;

  v8f acc[4] = {{}, {}, {}, {}};

#if USE_TDM
  if (wid == 0) {
    tdm_load_2d(&A[(size_t)row0 * Dc], lds_off(&ldsA[0][0]), 64, 128, Dc);
    tdm_load_2d(&Wt[(size_t)n0 * Dc], lds_off(&ldsB[0][0]), 64, 64, Dc);
  }
#endif

  for (int kk = 0, p = 0; kk < Dc; kk += 64, p ^= 1) {
#if USE_TDM
    if (wid == 0) __builtin_amdgcn_s_wait_tensorcnt(0);
    __syncthreads();  // data ready + other buffer free
    if (wid == 0 && kk + 64 < Dc) {
      tdm_load_2d(&A[(size_t)row0 * Dc + kk + 64], lds_off(&ldsA[p ^ 1][0]),
                  64, 128, Dc);
      tdm_load_2d(&Wt[(size_t)n0 * Dc + kk + 64], lds_off(&ldsB[p ^ 1][0]),
                  64, 64, Dc);
    }
#else
    __syncthreads();
    for (int u = tid; u < 1024; u += 256) {
      int r = u >> 3, seg = u & 7;
      *(float4*)&ldsA[p][r * 64 + seg * 8] =
          *(const float4*)&A[(size_t)(row0 + r) * Dc + kk + seg * 8];
    }
    for (int u = tid; u < 512; u += 256) {
      int r = u >> 3, seg = u & 7;
      *(float4*)&ldsB[p][r * 64 + seg * 8] =
          *(const float4*)&Wt[(size_t)(n0 + r) * Dc + kk + seg * 8];
    }
    __syncthreads();
#endif
    const __bf16* At = &ldsA[p][0];
    const __bf16* Bt = &ldsB[p][0];
    v16bf a0 = load_a_frag(At + wid * 16 * 64, 64);
    v16bf a1 = load_a_frag(At + wid * 16 * 64 + 32, 64);
    v16bf b00 = load_b_frag_nk(Bt + 0 * 16 * 64, 64);
    v16bf b01 = load_b_frag_nk(Bt + 0 * 16 * 64 + 32, 64);
    v16bf b10 = load_b_frag_nk(Bt + 1 * 16 * 64, 64);
    v16bf b11 = load_b_frag_nk(Bt + 1 * 16 * 64 + 32, 64);
    v16bf b20 = load_b_frag_nk(Bt + 2 * 16 * 64, 64);
    v16bf b21 = load_b_frag_nk(Bt + 2 * 16 * 64 + 32, 64);
    v16bf b30 = load_b_frag_nk(Bt + 3 * 16 * 64, 64);
    v16bf b31 = load_b_frag_nk(Bt + 3 * 16 * 64 + 32, 64);
    acc[0] = wmma_bf16(a0, b00, acc[0]);
    acc[1] = wmma_bf16(a0, b10, acc[1]);
    acc[2] = wmma_bf16(a0, b20, acc[2]);
    acc[3] = wmma_bf16(a0, b30, acc[3]);
    acc[0] = wmma_bf16(a1, b01, acc[0]);
    acc[1] = wmma_bf16(a1, b11, acc[1]);
    acc[2] = wmma_bf16(a1, b21, acc[2]);
    acc[3] = wmma_bf16(a1, b31, acc[3]);
  }

#pragma unroll
  for (int t = 0; t < 4; ++t) {
#pragma unroll
    for (int v = 0; v < 8; ++v) {
      int r = row0 + wid * 16 + v + 8 * sel;  // global row (b*S+s)
      int n = n0 + t * 16 + m16;              // global col (h*64+hd)
      float val = (acc[t][v] + bias[n]) * scale;
      if (mode == 2) {
        ((float*)outp)[(size_t)r * Dc + n] = val;
      } else {
        int bb = r >> 12, s = r & (Sc - 1);
        int hh = n >> 6, hd = n & 63;
        __bf16* ob = (__bf16*)outp;
        if (mode == 0)
          ob[(((size_t)bb * Hc + hh) * Sc + s) * HDc + hd] = (__bf16)val;
        else
          ob[(((size_t)bb * Hc + hh) * HDc + hd) * Sc + s] = (__bf16)val;
      }
    }
  }
}

// ----------------------------------------------------- banded flash attention
// grid (NC, H, B), 512 threads = 16 waves, wave w owns 16 query rows.
// idx 0 = global-key tile (cols < G valid), idx >= 1 = in-range window tiles.
__global__ __launch_bounds__(512) void band_attn_kernel(
    const __bf16* __restrict__ q,   // [B,H,S,HD]
    const __bf16* __restrict__ k,   // [B,H,S,HD]
    const __bf16* __restrict__ vt,  // [B,H,HD,S]
    const int* __restrict__ mask,   // [B,S] (<0 masked)
    __bf16* __restrict__ out)       // [B,S,D] merged bf16
{
  __shared__ __bf16 ldsK[2][32 * 64];      // [key][hd]
  __shared__ __bf16 ldsV[2][64 * 32];      // [hd][key]
  __shared__ __bf16 ldsP[16][16 * 32];     // per-wave P scratch

  int c = blockIdx.x, h = blockIdx.y, b = blockIdx.z;
  int tid = threadIdx.x;
  int wid = tid >> 5, lane = tid & 31;
  int m16 = lane & 15, sel = lane >> 4;

  const __bf16* qbh = q  + ((size_t)(b * Hc + h)) * Sc * HDc;
  const __bf16* kbh = k  + ((size_t)(b * Hc + h)) * Sc * HDc;
  const __bf16* vbh = vt + ((size_t)(b * Hc + h)) * HDc * Sc;
  const int*  maskb = mask + (size_t)b * Sc;

  // contiguous valid iteration range (tile edges align with 0/S)
  int kb_start = c * W1c - W1c;
  if (kb_start < 0) kb_start = 0;
  int kb_end = c * W1c + 2 * W1c;
  if (kb_end > Sc) kb_end = Sc;
  int nit = 1 + (kb_end - kb_start) / 32;

  int qrow0 = c * W1c + wid * 16;
  v16bf qa0 = load_a_frag(qbh + (size_t)qrow0 * HDc, HDc);
  v16bf qa1 = load_a_frag(qbh + (size_t)qrow0 * HDc + 32, HDc);

  float m8[8], l8[8];
  v8f O[4] = {{}, {}, {}, {}};
#pragma unroll
  for (int v = 0; v < 8; ++v) { m8[v] = -3.0e38f; l8[v] = 0.f; }

#if USE_TDM
  if (wid == 0) {  // idx 0 tile: global keys 0..31
    tdm_load_2d(kbh, lds_off(&ldsK[0][0]), 64, 32, 64);
    tdm_load_2d(vbh, lds_off(&ldsV[0][0]), 32, 64, Sc);
  }
#endif

  for (int idx = 0, p = 0; idx < nit; ++idx, p ^= 1) {
    int keyb = (idx == 0) ? 0 : kb_start + (idx - 1) * 32;

#if USE_TDM
    if (wid == 0) __builtin_amdgcn_s_wait_tensorcnt(0);
    __syncthreads();
    if (wid == 0 && idx + 1 < nit) {
      int nb = kb_start + idx * 32;
      tdm_load_2d(kbh + (size_t)nb * HDc, lds_off(&ldsK[p ^ 1][0]), 64, 32, 64);
      tdm_load_2d(vbh + nb, lds_off(&ldsV[p ^ 1][0]), 32, 64, Sc);
    }
#else
    __syncthreads();
    if (tid < 256) {  // K tile 32x64
      int key = tid >> 3, seg = tid & 7;
      *(float4*)&ldsK[p][key * 64 + seg * 8] =
          *(const float4*)&kbh[(size_t)(keyb + key) * HDc + seg * 8];
    } else {          // V tile 64x32 (transposed source)
      int u = tid - 256;
      int hd = u >> 2, seg = u & 3;
      *(float4*)&ldsV[p][hd * 32 + seg * 8] =
          *(const float4*)&vbh[(size_t)hd * Sc + keyb + seg * 8];
    }
    __syncthreads();
#endif

    const __bf16* Kt = &ldsK[p][0];
    const __bf16* Vt = &ldsV[p][0];

    // scores: S(16x32) = Q(16x64) x K^T
    v16bf kb00 = load_b_frag_nk(Kt + 0 * 64 + 0, 64);
    v16bf kb01 = load_b_frag_nk(Kt + 0 * 64 + 32, 64);
    v16bf kb10 = load_b_frag_nk(Kt + 16 * 64 + 0, 64);
    v16bf kb11 = load_b_frag_nk(Kt + 16 * 64 + 32, 64);
    v8f s0 = {}, s1 = {};
    s0 = wmma_bf16(qa0, kb00, s0);
    s1 = wmma_bf16(qa0, kb10, s1);
    s0 = wmma_bf16(qa1, kb01, s0);
    s1 = wmma_bf16(qa1, kb11, s1);

    // masking (mask reads hit L2; band test is pure ALU)
    int j0 = keyb + m16, j1 = keyb + 16 + m16;
    bool okc0, okc1;
    if (idx == 0) {
      okc0 = (m16 < Gc) && (maskb[m16] >= 0);
      okc1 = false;  // cols 16..31 >= G
    } else {
      okc0 = (maskb[j0] >= 0);
      okc1 = (maskb[j1] >= 0);
    }
#pragma unroll
    for (int v = 0; v < 8; ++v) {
      int i = qrow0 + v + 8 * sel;
      bool b0 = okc0, b1 = okc1;
      if (idx != 0) {
        b0 = b0 && (j0 - i <= W1c) && (i - j0 <= W1c);
        b1 = b1 && (j1 - i <= W1c) && (i - j1 <= W1c);
      }
      if (!b0) s0[v] = NEGF;
      if (!b1) s1[v] = NEGF;
    }

    flash_update(s0, s1, m8, l8, O, &ldsP[wid][0]);

    // O += P(16x32) x V(32x64)
    v16bf pa = load_a_frag(&ldsP[wid][0], 32);
    v16bf vb0 = load_b_frag_nk(Vt + 0 * 16 * 32, 32);
    v16bf vb1 = load_b_frag_nk(Vt + 1 * 16 * 32, 32);
    v16bf vb2 = load_b_frag_nk(Vt + 2 * 16 * 32, 32);
    v16bf vb3 = load_b_frag_nk(Vt + 3 * 16 * 32, 32);
    O[0] = wmma_bf16(pa, vb0, O[0]);
    O[1] = wmma_bf16(pa, vb1, O[1]);
    O[2] = wmma_bf16(pa, vb2, O[2]);
    O[3] = wmma_bf16(pa, vb3, O[3]);
  }

  // normalize + store merged; masked query rows -> 0
#pragma unroll
  for (int v = 0; v < 8; ++v) {
    int row = qrow0 + v + 8 * sel;
    float lv = l8[v];
    bool qneg = maskb[row] < 0;
    float inv = (lv > 0.f && !qneg) ? (1.f / lv) : 0.f;
#pragma unroll
    for (int t = 0; t < 4; ++t) {
      out[((size_t)b * Sc + row) * Dc + h * 64 + t * 16 + m16] =
          (__bf16)(O[t][v] * inv);
    }
  }
}

// ----------------------------------------------- global-token full attention
// grid (H, B), 256 threads = 8 waves; wave w handles keys [w*512, w*512+512).
__global__ __launch_bounds__(256) void global_attn_kernel(
    const __bf16* __restrict__ qg,   // [B,H,S,HD] (rows < G used)
    const __bf16* __restrict__ kg,   // [B,H,S,HD]
    const __bf16* __restrict__ vgt,  // [B,H,HD,S]
    const int* __restrict__ mask,    // [B,S]
    __bf16* __restrict__ out)        // [B,S,D] merged bf16 (rows < G)
{
  __shared__ float  ldsO[8][16][64];
  __shared__ float  ldsM[8][16];
  __shared__ float  ldsL[8][16];
  __shared__ __bf16 ldsP[8][16 * 32];

  int h = blockIdx.x, b = blockIdx.y;
  int tid = threadIdx.x;
  int wid = tid >> 5, lane = tid & 31;
  int m16 = lane & 15, sel = lane >> 4;

  const __bf16* qgbh = qg  + ((size_t)(b * Hc + h)) * Sc * HDc;
  const __bf16* kgbh = kg  + ((size_t)(b * Hc + h)) * Sc * HDc;
  const __bf16* vgbh = vgt + ((size_t)(b * Hc + h)) * HDc * Sc;
  const int*   maskb = mask + (size_t)b * Sc;

  v16bf qa0 = {}, qa1 = {};
  if (m16 < Gc) {
    const __bf16* r = qgbh + (size_t)m16 * HDc;
#pragma unroll
    for (int e = 0; e < 8; ++e) {
      qa0[e]     = r[8 * sel + e];
      qa0[8 + e] = r[16 + 8 * sel + e];
      qa1[e]     = r[32 + 8 * sel + e];
      qa1[8 + e] = r[48 + 8 * sel + e];
    }
  }

  float m8[8], l8[8];
  v8f O[4] = {{}, {}, {}, {}};
#pragma unroll
  for (int v = 0; v < 8; ++v) { m8[v] = -3.0e38f; l8[v] = 0.f; }

  for (int it = 0; it < 16; ++it) {
    int kb = wid * 512 + it * 32;
    v16bf kb00 = load_b_frag_nk(kgbh + (size_t)kb * HDc, HDc);
    v16bf kb01 = load_b_frag_nk(kgbh + (size_t)kb * HDc + 32, HDc);
    v16bf kb10 = load_b_frag_nk(kgbh + (size_t)(kb + 16) * HDc, HDc);
    v16bf kb11 = load_b_frag_nk(kgbh + (size_t)(kb + 16) * HDc + 32, HDc);
    v8f s0 = {}, s1 = {};
    s0 = wmma_bf16(qa0, kb00, s0);
    s1 = wmma_bf16(qa0, kb10, s1);
    s0 = wmma_bf16(qa1, kb01, s0);
    s1 = wmma_bf16(qa1, kb11, s1);

    bool okc0 = maskb[kb + m16] >= 0;
    bool okc1 = maskb[kb + 16 + m16] >= 0;
#pragma unroll
    for (int v = 0; v < 8; ++v) {
      if (!okc0) s0[v] = NEGF;
      if (!okc1) s1[v] = NEGF;
    }

    flash_update(s0, s1, m8, l8, O, &ldsP[wid][0]);

    v16bf pa = load_a_frag(&ldsP[wid][0], 32);
    v16bf vb0 = load_b_frag_nk(vgbh + (size_t)(0 * 16) * Sc + kb, Sc);
    v16bf vb1 = load_b_frag_nk(vgbh + (size_t)(1 * 16) * Sc + kb, Sc);
    v16bf vb2 = load_b_frag_nk(vgbh + (size_t)(2 * 16) * Sc + kb, Sc);
    v16bf vb3 = load_b_frag_nk(vgbh + (size_t)(3 * 16) * Sc + kb, Sc);
    O[0] = wmma_bf16(pa, vb0, O[0]);
    O[1] = wmma_bf16(pa, vb1, O[1]);
    O[2] = wmma_bf16(pa, vb2, O[2]);
    O[3] = wmma_bf16(pa, vb3, O[3]);
  }

#pragma unroll
  for (int t = 0; t < 4; ++t)
#pragma unroll
    for (int v = 0; v < 8; ++v)
      ldsO[wid][v + 8 * sel][t * 16 + m16] = O[t][v];
  if (m16 == 0) {
#pragma unroll
    for (int v = 0; v < 8; ++v) {
      ldsM[wid][v + 8 * sel] = m8[v];
      ldsL[wid][v + 8 * sel] = l8[v];
    }
  }
  __syncthreads();

  for (int e = tid; e < Gc * 64; e += 256) {
    int row = e >> 6, col = e & 63;
    float M = -3.0e38f;
#pragma unroll
    for (int w = 0; w < 8; ++w) M = fmaxf(M, ldsM[w][row]);
    float L = 0.f, accv = 0.f;
#pragma unroll
    for (int w = 0; w < 8; ++w) {
      float f = __expf(ldsM[w][row] - M);
      L += f * ldsL[w][row];
      accv += f * ldsO[w][row][col];
    }
    float r = (L > 0.f) ? (accv / L) : 0.f;
    out[((size_t)b * Sc + row) * Dc + h * 64 + col] = (__bf16)r;
  }
}

// ------------------------------------------------------------------- launch
extern "C" void kernel_launch(void* const* d_in, const int* in_sizes, int n_in,
                              void* d_out, int out_size, void* d_ws,
                              size_t ws_size, hipStream_t stream) {
  const float* hs   = (const float*)d_in[0];
  const int*   mask = (const int*)d_in[1];
  const float* W[7] = {(const float*)d_in[2],  (const float*)d_in[4],
                       (const float*)d_in[6],  (const float*)d_in[8],
                       (const float*)d_in[10], (const float*)d_in[12],
                       (const float*)d_in[14]};
  const float* bias[7] = {(const float*)d_in[3],  (const float*)d_in[5],
                          (const float*)d_in[7],  (const float*)d_in[9],
                          (const float*)d_in[11], (const float*)d_in[13],
                          (const float*)d_in[15]};

  __bf16* hsb   = (__bf16*)d_ws;           // [B*S, D]
  __bf16* wT    = hsb + BSDc;              // 7 x [N][K]
  __bf16* qb    = wT + 7 * WNc;            // [B,H,S,HD]
  __bf16* kbuf  = qb + BSDc;               // [B,H,S,HD]
  __bf16* vtb   = kbuf + BSDc;             // [B,H,HD,S]
  __bf16* qgb   = vtb + BSDc;              // [B,H,S,HD]
  __bf16* kgb   = qgb + BSDc;              // [B,H,S,HD]
  __bf16* vgtb  = kgb + BSDc;              // [B,H,HD,S]
  __bf16* attnm = vgtb + BSDc;             // [B,S,D] merged attention

  const float scaleQ = 0.125f;  // 1/sqrt(64)

  cvt_bf16_kernel<<<(int)(BSDc / 256), 256, 0, stream>>>(hs, hsb, (int)BSDc);
  for (int i = 0; i < 7; ++i)
    transpose_bf16_kernel<<<(int)(WNc / 256), 256, 0, stream>>>(W[i],
                                                                wT + i * WNc);

  dim3 gg(Bc * Sc / 128, Dc / 64);
  gemm_bf16_kernel<<<gg, 256, 0, stream>>>(hsb, wT + 0 * WNc, bias[0], qb,
                                           scaleQ, 0);  // q
  gemm_bf16_kernel<<<gg, 256, 0, stream>>>(hsb, wT + 1 * WNc, bias[1], kbuf,
                                           1.0f, 0);    // k
  gemm_bf16_kernel<<<gg, 256, 0, stream>>>(hsb, wT + 2 * WNc, bias[2], vtb,
                                           1.0f, 1);    // v (transposed)
  gemm_bf16_kernel<<<gg, 256, 0, stream>>>(hsb, wT + 3 * WNc, bias[3], qgb,
                                           scaleQ, 0);  // qg
  gemm_bf16_kernel<<<gg, 256, 0, stream>>>(hsb, wT + 4 * WNc, bias[4], kgb,
                                           1.0f, 0);    // kg
  gemm_bf16_kernel<<<gg, 256, 0, stream>>>(hsb, wT + 5 * WNc, bias[5], vgtb,
                                           1.0f, 1);    // vg (transposed)

  band_attn_kernel<<<dim3(NCc, Hc, Bc), 512, 0, stream>>>(qb, kbuf, vtb, mask,
                                                          attnm);
  global_attn_kernel<<<dim3(Hc, Bc), 256, 0, stream>>>(qgb, kgb, vgtb, mask,
                                                       attnm);

  gemm_bf16_kernel<<<gg, 256, 0, stream>>>(attnm, wT + 6 * WNc, bias[6], d_out,
                                           1.0f, 2);
}
